// CrossAttentionLayer_89644557402759
// MI455X (gfx1250) — matrix operand order repaired and checked by reading, not compiled, not run on hardware
//
#include <hip/hip_runtime.h>
#include <hip/hip_bf16.h>

// ---------------------------------------------------------------------------
// Cross-attention layer for MI455X (gfx1250, wave32, WMMA + TDM).
// Pipeline: qproj -> kvproj (TDM-staged) -> flash-attn (S^T = K*Q^T)
//           -> combine -> outproj+residual+LayerNorm.
// Softmax runs in exp2 domain: log2(e)/sqrt(d) folded into Q projection.
// ---------------------------------------------------------------------------

typedef __bf16 bf16_t;
typedef __attribute__((ext_vector_type(16))) __bf16 v16bf;
typedef __attribute__((ext_vector_type(8)))  __bf16 v8bf;
typedef __attribute__((ext_vector_type(8)))  float  v8f;
typedef __attribute__((ext_vector_type(4)))  float  v4f;
typedef unsigned int u32;
typedef __attribute__((ext_vector_type(4))) u32 u32x4;
typedef __attribute__((ext_vector_type(4))) int i32x4;
typedef __attribute__((ext_vector_type(8))) int i32x8;

#define Bb 2
#define Qq 256
#define Nn 16384
#define Cc 256
#define Hh 8
#define Dd 32
// (1/sqrt(32)) * log2(e): softmax evaluated with exp2.
#define QK_SCALE_L2E 0.2550348588f
#define NEG_BIG  (-1.0e9f)

union V16 { v16bf v; v8bf h[2]; bf16_t e[16]; };
union V8  { v8f v; float f[8]; };
union V8B { v8bf v; bf16_t e[8]; };

__device__ inline v8f wmma_bf16(v16bf a, v16bf b, v8f c) {
  return __builtin_amdgcn_wmma_f32_16x16x32_bf16(false, a, false, b, (short)0, c, false, false);
}
__device__ inline v8f zero8() { v8f z = {0.f,0.f,0.f,0.f,0.f,0.f,0.f,0.f}; return z; }

#if __has_builtin(__builtin_amdgcn_exp2f)
#define fast_exp2(x) __builtin_amdgcn_exp2f(x)
#else
#define fast_exp2(x) exp2f(x)
#endif

// ---------------------------------------------------------------------------
// TDM: load a 32(row) x 16(col) f32 tile (row stride = Nn elements) to LDS.
// D# group0: count=1, lds_addr, global_addr, type=2.
// D# group1: data_size=4B, tensor_dim0=16384, tensor_dim1=256,
//            tile_dim0=16, tile_dim1=32, tensor_dim0_stride=16384.
// ---------------------------------------------------------------------------
#if __has_builtin(__builtin_amdgcn_tensor_load_to_lds)
#define HAVE_TDM 1
__device__ inline void tdm_load_tile_32x16(void* lds_ptr, const float* gsrc) {
  u32 lds_off = (u32)__builtin_amdgcn_readfirstlane((int)(u32)(uintptr_t)lds_ptr);
  unsigned long long ga = (unsigned long long)(uintptr_t)gsrc;
  u32 galo = (u32)__builtin_amdgcn_readfirstlane((int)(u32)ga);
  u32 gahi = (u32)__builtin_amdgcn_readfirstlane((int)(u32)(ga >> 32));
  u32x4 g0 = { 1u, lds_off, galo, (gahi & 0x01FFFFFFu) | 0x80000000u };
  i32x8 g1 = { 0x00020000,        // data_size=4B (code 2) at bits 17:16
               0x40000000,        // tensor_dim0 = 16384 -> bits 63:48
               0x01000000,        // tensor_dim1 = 256   -> bits 95:80(lo16)
               0x00100000,        // tile_dim0 = 16      -> bits 127:112
               32,                // tile_dim1 = 32      -> bits 143:128
               16384,             // tensor_dim0_stride[31:0]
               0, 0 };
  i32x4 z4 = { 0, 0, 0, 0 };
#if defined(__clang_major__) && (__clang_major__ >= 23)
  i32x8 z8 = { 0, 0, 0, 0, 0, 0, 0, 0 };
  __builtin_amdgcn_tensor_load_to_lds(g0, g1, z4, z4, z8, 0);
#else
  __builtin_amdgcn_tensor_load_to_lds(g0, g1, z4, z4, 0);
#endif
}
#else
#define HAVE_TDM 0
#endif

// ---------------------------------------------------------------------------
// Kernel 1: Q projection. q = ((queries+query_embed) @ wq^T + bq) * scale*l2e
// Output qh bf16 [B][H][Q][32]; one wave per 16x16 tile: 512 waves.
// ---------------------------------------------------------------------------
__global__ void qproj_kernel(const float* __restrict__ queries,
                             const float* __restrict__ query_embed,
                             const float* __restrict__ in_proj_w,
                             const float* __restrict__ in_proj_b,
                             bf16_t* __restrict__ qh) {
  const int tid = threadIdx.x;
  const int lane = tid & 31, wave = tid >> 5;
  const int gw = blockIdx.x * 8 + wave;     // 0..511
  const int mtile = gw >> 4;                // 0..31 (rows of B*Q)
  const int ctile = gw & 15;                // 0..15 (output channels)
  const int half = lane >> 4;
  const int mrow = mtile * 16 + (lane & 15);  // A-operand row
  const int c    = ctile * 16 + (lane & 15);  // B-operand column

  const float* qrow = queries     + (size_t)mrow * Cc;
  const float* erow = query_embed + (size_t)mrow * Cc;
  const float* wrow = in_proj_w   + (size_t)c * Cc;   // wq rows [0,256)

  V8 acc; acc.v = zero8();
  for (int kk = 0; kk < Cc; kk += 32) {
    V16 a, b;
    const int d0 = kk + half * 8;
#pragma unroll
    for (int i = 0; i < 8; ++i) {
      a.e[i]     = (bf16_t)(qrow[d0 + i]      + erow[d0 + i]);
      a.e[8 + i] = (bf16_t)(qrow[d0 + 16 + i] + erow[d0 + 16 + i]);
    }
    const int k0 = kk + half * 8;
#pragma unroll
    for (int i = 0; i < 16; ++i) b.e[i] = (bf16_t)wrow[k0 + (i & 7) + ((i >> 3) << 4)];
    acc.v = wmma_bf16(a.v, b.v, acc.v);
  }

  const float bias = in_proj_b[c];
  const int h = c >> 5, d = c & 31;
#pragma unroll
  for (int r = 0; r < 8; ++r) {
    const int mr = mtile * 16 + half * 8 + r;   // D rows = (lane/16)*8 + r
    const int b0 = mr >> 8, qi = mr & 255;
    const float val = (acc.f[r] + bias) * QK_SCALE_L2E;
    qh[(((size_t)(b0 * Hh + h) * Qq) + qi) * Dd + d] = (bf16_t)val;
  }
}

// ---------------------------------------------------------------------------
// Kernel 2: K/V projections (n-major GEMM), activations staged via TDM.
//   k[n][c] = sum_c' (vf+pe)[c'][n] * wk[c][c'] + bk[c]
//   v[n][c] = sum_c'  vf[c'][n]     * wv[c][c'] + bv[c]
// kh bf16 [B][H][N][32]; vt bf16 [B][C][N].
// Wave = (b, n-tile, group of 4 c-tiles): 8192 waves, 8 waves/block.
// Each wave double-buffers two 32x16 f32 tiles (vf, pe) in private LDS.
// ---------------------------------------------------------------------------
__global__ void kvproj_kernel(const float* __restrict__ vid_features,
                              const float* __restrict__ pos_embed,
                              const float* __restrict__ in_proj_w,
                              const float* __restrict__ in_proj_b,
                              bf16_t* __restrict__ kh,
                              bf16_t* __restrict__ vt) {
  const int tid = threadIdx.x;
  const int lane = tid & 31, wave = tid >> 5;
  const int gw = blockIdx.x * 8 + wave;   // 0..8191
  const int b = gw >> 12;
  const int rem = gw & 4095;
  const int mtile = rem >> 2;             // 0..1023 n-tile
  const int cgroup = rem & 3;             // 0..3
  const int half = lane >> 4;
  const int x = lane & 15;
  const int n0 = mtile * 16;
  const int n = n0 + x;                   // A-operand row (position)

  const float* vfb = vid_features + (size_t)b * Cc * Nn;
  const float* peb = pos_embed    + (size_t)b * Cc * Nn;

  V8 acck[4], accv[4];
#pragma unroll
  for (int j = 0; j < 4; ++j) { acck[j].v = zero8(); accv[j].v = zero8(); }

#if HAVE_TDM
  // [wave][buf][vf|pe][32 rows * 16 cols]
  __shared__ float xt[8][2][2][32 * 16];
  tdm_load_tile_32x16(&xt[wave][0][0][0], vfb + n0);
  tdm_load_tile_32x16(&xt[wave][0][1][0], peb + n0);
  int cur = 0;
#endif

  for (int kk = 0; kk < Cc; kk += 32) {
    V16 av, ak;
#if HAVE_TDM
    if (kk + 32 < Cc) {   // prefetch next k-step into the other buffer
      tdm_load_tile_32x16(&xt[wave][cur ^ 1][0][0], vfb + (size_t)(kk + 32) * Nn + n0);
      tdm_load_tile_32x16(&xt[wave][cur ^ 1][1][0], peb + (size_t)(kk + 32) * Nn + n0);
      __builtin_amdgcn_s_wait_tensorcnt(2);
    } else {
      __builtin_amdgcn_s_wait_tensorcnt(0);
    }
    const float* bv_ = &xt[wave][cur][0][0];
    const float* bp_ = &xt[wave][cur][1][0];
#pragma unroll
    for (int i = 0; i < 8; ++i) {
      const int y1 = half * 8 + i;
      const int y2 = y1 + 16;
      const float v1 = bv_[y1 * 16 + x], p1 = bp_[y1 * 16 + x];
      const float v2 = bv_[y2 * 16 + x], p2 = bp_[y2 * 16 + x];
      av.e[i]     = (bf16_t)v1;        ak.e[i]     = (bf16_t)(v1 + p1);
      av.e[8 + i] = (bf16_t)v2;        ak.e[8 + i] = (bf16_t)(v2 + p2);
    }
    cur ^= 1;
#else
#pragma unroll
    for (int i = 0; i < 8; ++i) {
      const int c1 = kk + half * 8 + i;
      const int c2 = c1 + 16;
      const float v1 = vfb[(size_t)c1 * Nn + n];
      const float v2 = vfb[(size_t)c2 * Nn + n];
      const float p1 = peb[(size_t)c1 * Nn + n];
      const float p2 = peb[(size_t)c2 * Nn + n];
      av.e[i]     = (bf16_t)v1;        ak.e[i]     = (bf16_t)(v1 + p1);
      av.e[8 + i] = (bf16_t)v2;        ak.e[8 + i] = (bf16_t)(v2 + p2);
    }
#endif
#pragma unroll
    for (int j = 0; j < 4; ++j) {
      const int c = (cgroup * 4 + j) * 16 + (lane & 15);
      const float* wkrow = in_proj_w + (size_t)(Cc     + c) * Cc + kk + half * 8;
      const float* wvrow = in_proj_w + (size_t)(2 * Cc + c) * Cc + kk + half * 8;
      V16 bk, bv;
#pragma unroll
      for (int i = 0; i < 16; ++i) { bk.e[i] = (bf16_t)wkrow[(i & 7) + ((i >> 3) << 4)]; bv.e[i] = (bf16_t)wvrow[(i & 7) + ((i >> 3) << 4)]; }
      acck[j].v = wmma_bf16(ak.v, bk.v, acck[j].v);
      accv[j].v = wmma_bf16(av.v, bv.v, accv[j].v);
    }
  }

  const int nbase = mtile * 16 + half * 8;   // D rows
#pragma unroll
  for (int j = 0; j < 4; ++j) {
    const int c = (cgroup * 4 + j) * 16 + (lane & 15);
    const float bk = in_proj_b[Cc + c];
    const float bv = in_proj_b[2 * Cc + c];
    const int h = c >> 5, d = c & 31;
#pragma unroll
    for (int r = 0; r < 8; ++r)
      kh[(((size_t)(b * Hh + h) * Nn) + nbase + r) * Dd + d] = (bf16_t)(acck[j].f[r] + bk);
    V8B pack;
#pragma unroll
    for (int r = 0; r < 8; ++r) pack.e[r] = (bf16_t)(accv[j].f[r] + bv);
    *(v8bf*)(vt + ((size_t)(b * Cc + c)) * Nn + nbase) = pack.v;  // 16B store
  }
}

// ---------------------------------------------------------------------------
// Kernel 3: flash attention, transposed.  S^T = K * Q^T ; ctx^T = V^T * P^T.
// Softmax in exp2 domain; per-lane statistics (one pair shuffle per 32 keys).
// Wave = (b, h, q-tile, N-slice of 4096): 1024 waves. Partials to workspace.
// ---------------------------------------------------------------------------
__global__ void attn_kernel(const bf16_t* __restrict__ qh,
                            const bf16_t* __restrict__ kh,
                            const bf16_t* __restrict__ vt,
                            const float* __restrict__ vid_mask,
                            float* __restrict__ part) {
  const int tid = threadIdx.x;
  const int lane = tid & 31, wave = tid >> 5;
  const int gw = blockIdx.x * 8 + wave;    // 0..1023
  const int g = gw >> 2, slice = gw & 3;
  const int b = g >> 7, h = (g >> 4) & 7, qt = g & 15;
  const int half = lane >> 4;
  const int qloc = lane & 15;
  const int q = qt * 16 + qloc;

  // Q^T as B operand (loaded once): lane=q column, kk = half*8 + i (i<8), 16 + half*8 + (i-8) (i>=8).
  const bf16_t* qp = qh + (((size_t)(b * Hh + h) * Qq) + q) * Dd + half * 8;
  V16 bq; bq.h[0] = *(const v8bf*)qp; bq.h[1] = *(const v8bf*)(qp + 16);

  const bf16_t* kbase  = kh + ((size_t)(b * Hh + h) * Nn) * Dd;
  const bf16_t* v0base = vt + ((size_t)(b * Cc + h * Dd + qloc)) * Nn;       // d-tile 0
  const bf16_t* v1base = vt + ((size_t)(b * Cc + h * Dd + 16 + qloc)) * Nn;  // d-tile 1
  const float*  mrow   = vid_mask + ((size_t)(b * Qq + q)) * Nn;

  float mrun = -3.0e38f, lsum = 0.f;
  V8 acc0, acc1; acc0.v = zero8(); acc1.v = zero8();

  const int nend = (slice + 1) * (Nn / 4);
  for (int n0 = slice * (Nn / 4); n0 < nend; n0 += 32) {
    // K tiles as A operands (contiguous 16B chunks per lane).
    const bf16_t* k0p = kbase + ((size_t)(n0 + qloc)) * Dd + half * 8;
    const bf16_t* k1p = kbase + ((size_t)(n0 + 16 + qloc)) * Dd + half * 8;
    V16 ak0, ak1;
    ak0.h[0] = *(const v8bf*)k0p; ak0.h[1] = *(const v8bf*)(k0p + 16);
    ak1.h[0] = *(const v8bf*)k1p; ak1.h[1] = *(const v8bf*)(k1p + 16);

    V8 s0, s1;
    s0.v = wmma_bf16(ak0.v, bq.v, zero8());   // keys n0..n0+15   (rows), col=q
    s1.v = wmma_bf16(ak1.v, bq.v, zero8());   // keys n0+16..+31

    // Hard mask (q fixed per lane, keys contiguous -> float4 loads).
    const float* mp = mrow + n0 + half * 8;
    v4f ma = *(const v4f*)(mp);      v4f mb = *(const v4f*)(mp + 4);
    v4f mc = *(const v4f*)(mp + 16); v4f md = *(const v4f*)(mp + 20);
#pragma unroll
    for (int r = 0; r < 4; ++r) {
      if (ma[r] < 0.5f) s0.f[r]     = NEG_BIG;
      if (mb[r] < 0.5f) s0.f[4 + r] = NEG_BIG;
      if (mc[r] < 0.5f) s1.f[r]     = NEG_BIG;
      if (md[r] < 0.5f) s1.f[4 + r] = NEG_BIG;
    }

    // Online softmax (exp2 domain): per-lane max + one pair shuffle.
    float lm = s0.f[0];
#pragma unroll
    for (int r = 1; r < 8; ++r) lm = fmaxf(lm, s0.f[r]);
#pragma unroll
    for (int r = 0; r < 8; ++r) lm = fmaxf(lm, s1.f[r]);
    const float pm = fmaxf(lm, __shfl_xor(lm, 16, 32));
    const float mnew = fmaxf(mrun, pm);
    const float corr = fast_exp2(mrun - mnew);
    mrun = mnew;
    acc0.v = acc0.v * corr;
    acc1.v = acc1.v * corr;
    lsum *= corr;

    V8 p0, p1;
    float psum = 0.f;
#pragma unroll
    for (int r = 0; r < 8; ++r) {
      p0.f[r] = fast_exp2(s0.f[r] - mnew);
      p1.f[r] = fast_exp2(s1.f[r] - mnew);
      psum += p0.f[r] + p1.f[r];
    }
    lsum += psum;

    // Assemble P^T B-operand from the D-layout via one pair shuffle per reg.
    V16 bp;
#pragma unroll
    for (int r = 0; r < 8; ++r) {
      const float send = half ? p0.f[r] : p1.f[r];
      const float recv = __shfl_xor(send, 16, 32);
      bp.e[r]     = (bf16_t)(half ? recv    : p0.f[r]);
      bp.e[8 + r] = (bf16_t)(half ? p1.f[r] : recv);
    }

    // V^T tiles as A operands (contiguous along keys).
    const bf16_t* vp0 = v0base + n0 + half * 16;
    const bf16_t* vp1 = v1base + n0 + half * 16;
    V16 av0, av1;
    av0.h[0] = *(const v8bf*)vp0; av0.h[1] = *(const v8bf*)(vp0 + 8);
    av1.h[0] = *(const v8bf*)vp1; av1.h[1] = *(const v8bf*)(vp1 + 8);
    acc0.v = wmma_bf16(av0.v, bp.v, acc0.v);
    acc1.v = wmma_bf16(av1.v, bp.v, acc1.v);

    if (n0 + 32 < nend) {  // global_prefetch_b8 for next K/V tiles
      __builtin_prefetch(kbase + ((size_t)(n0 + 32 + qloc)) * Dd, 0, 1);
      __builtin_prefetch(v0base + n0 + 32, 0, 1);
    }
  }

  const float ltot = lsum + __shfl_xor(lsum, 16, 32);
  float* pb = part + ((size_t)(g * 4 + slice)) * 544;
#pragma unroll
  for (int r = 0; r < 8; ++r) {
    pb[qloc * 32 + half * 8 + r]      = acc0.f[r];   // d = half*8+r
    pb[qloc * 32 + 16 + half * 8 + r] = acc1.f[r];   // d = 16+half*8+r
  }
  if (half == 0) { pb[512 + qloc] = mrun; pb[528 + qloc] = ltot; }
}

// ---------------------------------------------------------------------------
// Kernel 3b: merge the 4 flash partials per (b,h,q-tile). (exp2 domain)
// ---------------------------------------------------------------------------
__global__ void combine_kernel(const float* __restrict__ part,
                               float* __restrict__ ctxbuf) {
  const int g = blockIdx.x;    // 0..255
  const int t = threadIdx.x;   // 0..31
  const int b = g >> 7, h = (g >> 4) & 7, qt = g & 15;
  const float* p0 = part + (size_t)g * 4 * 544;
  for (int idx = t; idx < 512; idx += 32) {
    const int ql = idx >> 5, d = idx & 31;
    float M = -3.0e38f;
#pragma unroll
    for (int s = 0; s < 4; ++s) M = fmaxf(M, p0[s * 544 + 512 + ql]);
    float L = 0.f, num = 0.f;
#pragma unroll
    for (int s = 0; s < 4; ++s) {
      const float w = fast_exp2(p0[s * 544 + 512 + ql] - M);
      L   += w * p0[s * 544 + 528 + ql];
      num += w * p0[s * 544 + ql * 32 + d];
    }
    ctxbuf[(((size_t)(b * Qq + qt * 16 + ql)) * Cc) + h * 32 + d] = num / L;
  }
}

// ---------------------------------------------------------------------------
// Kernel 4: out-proj + bias + residual + LayerNorm. One block per row.
// ---------------------------------------------------------------------------
__global__ void outln_kernel(const float* __restrict__ ctxbuf,
                             const float* __restrict__ queries,
                             const float* __restrict__ out_w,
                             const float* __restrict__ out_b,
                             const float* __restrict__ gamma,
                             const float* __restrict__ beta,
                             float* __restrict__ out) {
  const int row = blockIdx.x;   // 0..511
  const int c = threadIdx.x;    // 0..255
  __shared__ float ctxs[Cc];
  __shared__ float red[Cc];
  ctxs[c] = ctxbuf[(size_t)row * Cc + c];
  __syncthreads();
  const float* w = out_w + (size_t)c * Cc;
  float dot = 0.f;
#pragma unroll 8
  for (int k = 0; k < Cc; ++k) dot += ctxs[k] * w[k];
  const float x = queries[(size_t)row * Cc + c] + dot + out_b[c];

  red[c] = x; __syncthreads();
  for (int s = 128; s > 0; s >>= 1) { if (c < s) red[c] += red[c + s]; __syncthreads(); }
  const float mu = red[0] * (1.0f / Cc);
  __syncthreads();
  const float xc = x - mu;
  red[c] = xc * xc; __syncthreads();
  for (int s = 128; s > 0; s >>= 1) { if (c < s) red[c] += red[c + s]; __syncthreads(); }
  const float var = red[0] * (1.0f / Cc);
  out[(size_t)row * Cc + c] = xc * rsqrtf(var + 1e-5f) * gamma[c] + beta[c];
}

// ---------------------------------------------------------------------------
// Workspace layout (bytes):
//   qh   @ 0        : 262144   (bf16 [2][8][256][32])
//   kh   @ 262144   : 16777216 (bf16 [2][8][16384][32])
//   vt   @ 17039360 : 16777216 (bf16 [2][256][16384])
//   part @ 33816576 : 2228224  (f32, 1024 partial blocks of 544)
//   ctx  @ 36044800 : 524288   (f32 [2][256][256])
// ---------------------------------------------------------------------------
extern "C" void kernel_launch(void* const* d_in, const int* in_sizes, int n_in,
                              void* d_out, int out_size, void* d_ws, size_t ws_size,
                              hipStream_t stream) {
  const float* queries   = (const float*)d_in[0];
  const float* vid_feat  = (const float*)d_in[1];
  const float* vid_mask  = (const float*)d_in[2];
  const float* pos_embed = (const float*)d_in[3];
  const float* query_emb = (const float*)d_in[4];
  const float* in_w      = (const float*)d_in[5];
  const float* in_b      = (const float*)d_in[6];
  const float* out_w     = (const float*)d_in[7];
  const float* out_b     = (const float*)d_in[8];
  const float* gamma     = (const float*)d_in[9];
  const float* beta      = (const float*)d_in[10];

  char* ws = (char*)d_ws;
  bf16_t* qh     = (bf16_t*)(ws);
  bf16_t* kh     = (bf16_t*)(ws + 262144);
  bf16_t* vt     = (bf16_t*)(ws + 17039360);
  float*  part   = (float*)(ws + 33816576);
  float*  ctxbuf = (float*)(ws + 36044800);

  qproj_kernel <<<  64, 256, 0, stream>>>(queries, query_emb, in_w, in_b, qh);
  kvproj_kernel<<<1024, 256, 0, stream>>>(vid_feat, pos_embed, in_w, in_b, kh, vt);
  attn_kernel  <<< 128, 256, 0, stream>>>(qh, kh, vt, vid_mask, part);
  combine_kernel<<<256,  32, 0, stream>>>(part, ctxbuf);
  outln_kernel <<< 512, 256, 0, stream>>>(ctxbuf, queries, out_w, out_b, gamma, beta,
                                          (float*)d_out);
}
